// RGCNNoReLU_38225208934410
// MI455X (gfx1250) — matrix operand order repaired and checked by reading, not compiled, not run on hardware
//
#include <hip/hip_runtime.h>
#include <hip/hip_bf16.h>

// ---------------- problem constants (match reference) ----------------
#define NF 100000
#define NA 4000
#define NN 104000          // NF + NA
#define H  256
#define LAYERS 4
#define RR 4
#define BB 3
#define E_FA 200000
#define E_AF 200000
#define E_FF 400000
#define E_AA 40000

#define KBLK 5             // 4 relations + root  -> fused K = 5*H = 1280
#define LDS_ROW (KBLK * H + 8)   // halves per LDS row, +8 halves pad (16B) kills bank conflicts

typedef _Float16 v8h  __attribute__((ext_vector_type(8)));
typedef _Float16 v16h __attribute__((ext_vector_type(16)));
typedef float    v8f  __attribute__((ext_vector_type(8)));

#define ATOM_ADD(p, v) __hip_atomic_fetch_add((p), (v), __ATOMIC_RELAXED, __HIP_MEMORY_SCOPE_AGENT)

// CDNA5 async global->LDS copy: each lane copies 16B; tracked by ASYNCcnt.
__device__ __forceinline__ void async_copy_b128(unsigned lds_byte_addr, const _Float16* gaddr) {
    asm volatile("global_load_async_to_lds_b128 %0, %1, off"
                 :: "v"(lds_byte_addr), "v"(gaddr)
                 : "memory");
}
__device__ __forceinline__ void async_wait0() {
    asm volatile("s_wait_asynccnt 0" ::: "memory");
}

// ---------------- utility kernels ----------------
__global__ __launch_bounds__(256) void zero_f32(float* __restrict__ p, int n) {
    int i = blockIdx.x * 256 + threadIdx.x;
    if (i < n) p[i] = 0.0f;
}

// in-degree count per (relation, dst)
__global__ __launch_bounds__(256) void count_kernel(const int* __restrict__ dst, int doff,
                                                    int E, float* __restrict__ cntr) {
    int e = blockIdx.x * 256 + threadIdx.x;
    if (e < E) ATOM_ADD(&cntr[dst[e] + doff], 1.0f);
}

__global__ __launch_bounds__(256) void rden_kernel(const float* __restrict__ cnt,
                                                   float* __restrict__ rden, int n) {
    int i = blockIdx.x * 256 + threadIdx.x;
    if (i < n) rden[i] = 1.0f / fmaxf(cnt[i], 1.0f);
}

// ---------------- encoder: per-type Linear into x (f32) and xh (f16 slice of ABig) ----------------
__global__ __launch_bounds__(256) void encoder_kernel(const float* __restrict__ xf,
                                                      const float* __restrict__ xa,
                                                      const float* __restrict__ wf,
                                                      const float* __restrict__ bf,
                                                      const float* __restrict__ wa,
                                                      const float* __restrict__ ba,
                                                      float* __restrict__ x,
                                                      _Float16* __restrict__ xh) {
    int tid = blockIdx.x * 256 + threadIdx.x;     // < NN*H
    int n = tid >> 8;
    int c = tid & 255;
    float acc;
    if (n < NF) {
        acc = bf[c];
        const float* xr = xf + n * 32;
#pragma unroll
        for (int k = 0; k < 32; ++k) acc += xr[k] * wf[k * H + c];
    } else {
        int m = n - NF;
        acc = ba[c];
        const float* xr = xa + m * 16;
#pragma unroll
        for (int k = 0; k < 16; ++k) acc += xr[k] * wa[k * H + c];
    }
    x[tid]  = acc;
    xh[tid] = (_Float16)acc;
}

// ---------------- per-layer weight build: Wt[r][o][k] (transposed, f16) ----------------
// r<4: W_r = sum_b comp[r,b] * basis[b]  (basis layout [b][k][o]); r==4: root ([k][o])
__global__ __launch_bounds__(256) void build_weights(const float* __restrict__ basis_l,
                                                     const float* __restrict__ comp_l,
                                                     const float* __restrict__ root_l,
                                                     _Float16* __restrict__ Wt) {
    int tid = blockIdx.x * 256 + threadIdx.x;     // < 5*H*H
    int r = tid >> 16;
    int o = (tid >> 8) & 255;
    int k = tid & 255;
    float w;
    if (r < RR) {
        w = 0.0f;
#pragma unroll
        for (int b = 0; b < BB; ++b)
            w += comp_l[r * BB + b] * basis_l[(b * H + k) * H + o];
    } else {
        w = root_l[k * H + o];
    }
    Wt[tid] = (_Float16)w;                        // tid == r*65536 + o*256 + k
}

// ---------------- edge scatter: agg[dst] += x[src] (one relation) ----------------
// 64 threads per edge, float4 per thread, f32 global atomics
__global__ __launch_bounds__(256) void scatter_kernel(const int* __restrict__ src, int soff,
                                                      const int* __restrict__ dst, int doff,
                                                      int E,
                                                      const float* __restrict__ x,
                                                      float* __restrict__ agg) {
    unsigned tid = blockIdx.x * 256u + threadIdx.x;
    unsigned e = tid >> 6;
    if (e >= (unsigned)E) return;
    int c = (int)(tid & 63u) << 2;
    int s = src[e] + soff;
    int d = dst[e] + doff;
    const float4 v = *(const float4*)(x + s * H + c);
    float* p = agg + d * H + c;
    ATOM_ADD(p + 0, v.x);
    ATOM_ADD(p + 1, v.y);
    ATOM_ADD(p + 2, v.z);
    ATOM_ADD(p + 3, v.w);
}

// Ah_r = f16(agg * 1/deg) into relation slice of ABig; re-zero agg for next relation
__global__ __launch_bounds__(256) void convert_agg(float* __restrict__ agg,
                                                   const float* __restrict__ rden_r,
                                                   _Float16* __restrict__ Ah) {
    int tid = blockIdx.x * 256 + threadIdx.x;     // < NN*H
    int n = tid >> 8;
    float a = agg[tid] * rden_r[n];
    Ah[tid] = (_Float16)a;
    agg[tid] = 0.0f;
}

// ---------------- fused WMMA GEMM: C[M,256] = bias + ABig[M, 5*256] @ W[5*256, 256] ----------------
// ABig = 5 contiguous [NN][H] f16 blocks (4 relation aggregates + x). Wt[r][o][k] f16.
// Block = 512 threads = 16 waves; one 16-row tile per block, wave tn covers columns tn*16..+15.
// A tile (16 x 1280 halves = 40KB) staged in LDS once per block via async global->LDS copies.
__global__ __launch_bounds__(512) void wmma_gemm_fused(const _Float16* __restrict__ ABig,
                                                       const _Float16* __restrict__ Wt,
                                                       const float* __restrict__ bias_l,
                                                       float* __restrict__ C) {
    __shared__ _Float16 smem[16 * LDS_ROW];       // 41,216 B
    const int tm = blockIdx.x;                    // row tile (NN/16 blocks)
    const int t  = threadIdx.x;
    const unsigned ldsbase = (unsigned)(uintptr_t)(&smem[0]);

    // ---- stage A tile into LDS: 2560 x 16B chunks, 5 per thread, coalesced per (r,row) ----
#pragma unroll
    for (int i = 0; i < 5; ++i) {
        int c   = t + i * 512;                    // 0..2559
        int r   = c >> 9;                         // 512 chunks per K-block
        int row = (c >> 5) & 15;
        int kc  = c & 31;                         // 16B chunk within row segment
        const _Float16* g = ABig + ((size_t)r * NN + (size_t)tm * 16 + row) * H + kc * 8;
        unsigned lds = ldsbase + (unsigned)(row * (LDS_ROW * 2) + (r * H + kc * 8) * 2);
        async_copy_b128(lds, g);
    }
    async_wait0();
    __syncthreads();

    // ---- compute ----
    const int lane = t & 31;
    const int tn   = t >> 5;                      // 0..15 column tile
    const int l15  = lane & 15;
    const bool hi  = lane >= 16;
    const int col  = tn * 16 + l15;

    // A fragment source in LDS (16x32 A, contiguous k order, B uses the same):
    // lanes 0-15: row=lane,   K 0..15;  lanes 16-31: row=lane-16, K 16..31
    const _Float16* ap = &smem[l15 * LDS_ROW + (hi ? 16 : 0)];
    // B fragment (32x16): lanes 0-15 col=lane K 0..15; lanes 16-31 col=lane-16 K 16..31
    const _Float16* bcol = Wt + (size_t)col * H + (hi ? 16 : 0);

    float bias = bias_l[col];
    v8f acc;
#pragma unroll
    for (int v = 0; v < 8; ++v) acc[v] = bias;

#pragma unroll
    for (int r = 0; r < KBLK; ++r) {
        const _Float16* ar = ap + r * H;
        const _Float16* br = bcol + (size_t)r * H * H;
#pragma unroll
        for (int k0 = 0; k0 < H; k0 += 32) {
            v8h alo = *(const v8h*)(ar + k0);
            v8h ahi = *(const v8h*)(ar + k0 + 8);
            v16h a = __builtin_shufflevector(alo, ahi,
                                             0, 1, 2, 3, 4, 5, 6, 7,
                                             8, 9, 10, 11, 12, 13, 14, 15);
            v16h b = *(const v16h*)(br + k0);
            acc = __builtin_amdgcn_wmma_f32_16x16x32_f16(
                /*neg_a=*/false, a, /*neg_b=*/false, b,
                /*c_mod=*/(short)0, acc, /*reuse_a=*/false, /*reuse_b=*/false);
        }
    }

    // C/D layout: VGPR v -> row tm*16 + v + (hi?8:0), col
    float* cptr = C + ((size_t)tm * 16 + (hi ? 8 : 0)) * H + col;
#pragma unroll
    for (int v = 0; v < 8; ++v) cptr[v * H] = acc[v];
}

// ---------------- LayerNorm + leaky_relu(0.1) + residual; wave per row ----------------
__global__ __launch_bounds__(256) void ln_epilogue(const float* __restrict__ h,
                                                   float* __restrict__ x,
                                                   _Float16* __restrict__ xh,
                                                   const float* __restrict__ g,
                                                   const float* __restrict__ bta) {
    int lane = threadIdx.x & 31;
    int row = blockIdx.x * 8 + (threadIdx.x >> 5);
    if (row >= NN) return;
    const float* hp = h + (size_t)row * H + lane * 8;
    float v[8];
    float s = 0.0f, ss = 0.0f;
#pragma unroll
    for (int i = 0; i < 8; ++i) { v[i] = hp[i]; s += v[i]; ss += v[i] * v[i]; }
#pragma unroll
    for (int m = 16; m >= 1; m >>= 1) {
        s  += __shfl_xor(s,  m, 32);
        ss += __shfl_xor(ss, m, 32);
    }
    float mu   = s * (1.0f / H);
    float var  = ss * (1.0f / H) - mu * mu;
    float rstd = rsqrtf(var + 1e-5f);
    float* xp = x + (size_t)row * H + lane * 8;
    _Float16* xhp = xh + (size_t)row * H + lane * 8;
#pragma unroll
    for (int i = 0; i < 8; ++i) {
        int c = lane * 8 + i;
        float y = (v[i] - mu) * rstd * g[c] + bta[c];
        y = (y > 0.0f) ? y : 0.1f * y;
        float xn = y + xp[i];
        xp[i]  = xn;
        xhp[i] = (_Float16)xn;
    }
}

// ---------------- readout over flight rows; wave per row ----------------
__global__ __launch_bounds__(256) void readout_kernel(const float* __restrict__ x,
                                                      const float* __restrict__ ro_w,
                                                      const float* __restrict__ ro_b,
                                                      float* __restrict__ out) {
    int lane = threadIdx.x & 31;
    int row = blockIdx.x * 8 + (threadIdx.x >> 5);
    if (row >= NF) return;
    const float* xp = x + (size_t)row * H + lane * 8;
    const float* wp = ro_w + lane * 8;
    float s = 0.0f;
#pragma unroll
    for (int i = 0; i < 8; ++i) s += xp[i] * wp[i];
#pragma unroll
    for (int m = 16; m >= 1; m >>= 1) s += __shfl_xor(s, m, 32);
    if (lane == 0) out[row] = s + ro_b[0];
}

// ---------------- host-side orchestration ----------------
extern "C" void kernel_launch(void* const* d_in, const int* in_sizes, int n_in,
                              void* d_out, int out_size, void* d_ws, size_t ws_size,
                              hipStream_t stream) {
    (void)in_sizes; (void)n_in; (void)out_size; (void)ws_size;
    const float* x_flight  = (const float*)d_in[0];
    const float* x_airport = (const float*)d_in[1];
    const float* enc_w_f   = (const float*)d_in[2];
    const float* enc_b_f   = (const float*)d_in[3];
    const float* enc_w_a   = (const float*)d_in[4];
    const float* enc_b_a   = (const float*)d_in[5];
    const float* basis     = (const float*)d_in[6];
    const float* comp      = (const float*)d_in[7];
    const float* root      = (const float*)d_in[8];
    const float* conv_bias = (const float*)d_in[9];
    const float* ln_gamma  = (const float*)d_in[10];
    const float* ln_beta   = (const float*)d_in[11];
    const float* ro_w      = (const float*)d_in[12];
    const float* ro_b      = (const float*)d_in[13];
    const int* ei_fa = (const int*)d_in[14];
    const int* ei_af = (const int*)d_in[15];
    const int* ei_ff = (const int*)d_in[16];
    const int* ei_aa = (const int*)d_in[17];
    float* out = (float*)d_out;

    // workspace carve-up
    char* ws = (char*)d_ws;
    const size_t SZF  = (size_t)NN * H * 4;       // 106,496,000 B
    const size_t SZH2 = (size_t)NN * H * 2;
    float*    x    = (float*)(ws);
    float*    h    = (float*)(ws + SZF);
    float*    agg  = (float*)(ws + 2 * SZF);
    _Float16* ABig = (_Float16*)(ws + 3 * SZF);   // [5][NN][H] f16: rel0..3 aggregates, then xh
    float*    cnt  = (float*)(ws + 3 * SZF + KBLK * SZH2);
    float*    rden = (float*)(ws + 3 * SZF + KBLK * SZH2 + (size_t)RR * NN * 4);
    _Float16* Wt   = (_Float16*)(ws + 3 * SZF + KBLK * SZH2 + 2 * (size_t)RR * NN * 4);
    _Float16* xh   = ABig + (size_t)RR * NN * H;  // x slice of the fused A matrix

    const int NH_BLOCKS = (NN * H) / 256;         // 104000

    // relation table: {src_row, src_off, dst_row, dst_off, E}
    struct Rel { const int* s; int so; const int* d; int dof; int E; };
    const Rel rel[RR] = {
        { ei_fa, 0,  ei_fa + E_FA, NF, E_FA },
        { ei_af, NF, ei_af + E_AF, 0,  E_AF },
        { ei_ff, 0,  ei_ff + E_FF, 0,  E_FF },
        { ei_aa, NF, ei_aa + E_AA, NF, E_AA },
    };

    // 1) per-(relation,dst) in-degree -> reciprocal (layer-invariant)
    zero_f32<<<(RR * NN + 255) / 256, 256, 0, stream>>>(cnt, RR * NN);
    for (int r = 0; r < RR; ++r)
        count_kernel<<<(rel[r].E + 255) / 256, 256, 0, stream>>>(rel[r].d, rel[r].dof,
                                                                 rel[r].E, cnt + r * NN);
    rden_kernel<<<(RR * NN + 255) / 256, 256, 0, stream>>>(cnt, rden, RR * NN);

    // 2) encoders -> x (f32) + xh (f16 slice)
    encoder_kernel<<<NH_BLOCKS, 256, 0, stream>>>(x_flight, x_airport, enc_w_f, enc_b_f,
                                                  enc_w_a, enc_b_a, x, xh);

    // 3) zero the relation aggregation buffer once (convert_agg re-zeroes it)
    zero_f32<<<NH_BLOCKS, 256, 0, stream>>>(agg, NN * H);

    // 4) RGCN layers
    for (int l = 0; l < LAYERS; ++l) {
        build_weights<<<(KBLK * H * H) / 256, 256, 0, stream>>>(
            basis + (size_t)l * BB * H * H, comp + l * RR * BB,
            root + (size_t)l * H * H, Wt);

        for (int r = 0; r < RR; ++r) {
            scatter_kernel<<<rel[r].E / 4, 256, 0, stream>>>(rel[r].s, rel[r].so,
                                                             rel[r].d, rel[r].dof,
                                                             rel[r].E, x, agg);
            convert_agg<<<NH_BLOCKS, 256, 0, stream>>>(agg, rden + r * NN,
                                                       ABig + (size_t)r * NN * H);
        }

        // fused 5-block GEMM with bias init; h written exactly once
        wmma_gemm_fused<<<NN / 16, 512, 0, stream>>>(ABig, Wt, conv_bias + l * H, h);

        ln_epilogue<<<NN / 8, 256, 0, stream>>>(h, x, xh,
                                                ln_gamma + l * H, ln_beta + l * H);
    }

    // 5) readout
    readout_kernel<<<(NF + 7) / 8, 256, 0, stream>>>(x, ro_w, ro_b, out);
}
